// MultiHeadSelfAttentionEinSum2D_14912126451900
// MI455X (gfx1250) — matrix operand order repaired and checked by reading, not compiled, not run on hardware
//
#include <hip/hip_runtime.h>
#include <math.h>
#include <stdint.h>

// ---------------------------------------------------------------------------
// MI455X (gfx1250) multi-head self-attention, bf16 WMMA pipeline + TDM.
//   x[8,4,1024,384] @ w_qkv[384,8,144] -> Q,K,V (PD=48 padded to 64)
//   flash attention per (b,p,h), N=1024, K/V tiles streamed to LDS by the
//   Tensor Data Mover (double buffered, TENSORcnt-tracked)
//   out = wv @ w_o[48,8,384] + b_o   (fp32 output)
// ---------------------------------------------------------------------------

#define B_   8
#define P_   4
#define N_   1024
#define D_   384
#define H_   8
#define PD_  48
#define PDP  64      // PD padded to multiple of 32 for WMMA K-chunks
#define C3   1152    // H * 3 * PD
#define KOUT 512     // H * PDP
#define INV_SCALE 0.14433756729740643f   // 1/sqrt(48)

typedef __bf16 bf16;
typedef __attribute__((ext_vector_type(8)))  __bf16 bf16x8;
typedef __attribute__((ext_vector_type(16))) __bf16 bf16x16;
typedef __attribute__((ext_vector_type(8)))  float  f32x8;
typedef __attribute__((ext_vector_type(4)))  unsigned int u32x4;
typedef __attribute__((ext_vector_type(8)))  int i32x8;
typedef __attribute__((ext_vector_type(4)))  int i32x4;

#if __has_builtin(__builtin_amdgcn_tensor_load_to_lds)
#define HAS_TDM 1
#else
#define HAS_TDM 0
#endif

// Workspace layout (bytes)
static constexpr size_t SZ_XB   = (size_t)B_ * P_ * N_ * D_ * 2;      // 25,165,824
static constexpr size_t SZ_WQT  = (size_t)C3 * D_ * 2;                //    884,736
static constexpr size_t SZ_WOT  = (size_t)D_ * KOUT * 2;              //    393,216
static constexpr size_t SZ_QKV1 = (size_t)B_ * P_ * H_ * N_ * PDP * 2;// 33,554,432
static constexpr size_t OFF_XB  = 0;
static constexpr size_t OFF_WQT = OFF_XB  + SZ_XB;
static constexpr size_t OFF_WOT = OFF_WQT + SZ_WQT;
static constexpr size_t OFF_Q   = OFF_WOT + SZ_WOT;
static constexpr size_t OFF_K   = OFF_Q   + SZ_QKV1;
static constexpr size_t OFF_VT  = OFF_K   + SZ_QKV1;
static constexpr size_t OFF_WV  = OFF_VT  + SZ_QKV1;

__device__ __forceinline__ f32x8 wmma_bf16(bf16x16 a, bf16x16 b, f32x8 c) {
  // D = A(16x32 bf16) * B(32x16 bf16) + C(16x16 f32)
  return __builtin_amdgcn_wmma_f32_16x16x32_bf16(false, a, false, b,
                                                 (short)0, c, false, false);
}

// A-matrix 16x32 bf16 per-lane layout (ISA 7.12.2): lane l holds row M=l%16,
// K elements { kb..kb+7, kb+16..kb+23 } with kb = 8*(l/16).
__device__ __forceinline__ bf16x16 load_fragA(const bf16* p, int lane) {
  const int kb = (lane >> 4) * 8;
  bf16x8 lo = *(const bf16x8*)(p + kb);
  bf16x8 hi = *(const bf16x8*)(p + kb + 16);
  bf16x16 r;
#pragma unroll
  for (int i = 0; i < 8; ++i) { r[i] = lo[i]; r[i + 8] = hi[i]; }
  return r;
}

// B-matrix 32x16 bf16 per-lane layout: lane l holds col N=l%16,
// K elements { kb..kb+7, kb+16..kb+23 } with kb = 8*(l/16).
__device__ __forceinline__ bf16x16 load_fragB(const bf16* p, int lane) {
  const int kb = (lane >> 4) * 8;
  bf16x8 lo = *(const bf16x8*)(p + kb);
  bf16x8 hi = *(const bf16x8*)(p + kb + 16);
  bf16x16 r;
#pragma unroll
  for (int i = 0; i < 8; ++i) { r[i] = lo[i]; r[i + 8] = hi[i]; }
  return r;
}

#if HAS_TDM
__device__ __forceinline__ int rfl(unsigned v) {
  return __builtin_amdgcn_readfirstlane((int)v);
}

// TDM 2-D tile load: tile_dim0=64 elems (128B rows), tile_dim1=64 rows,
// data_size=2B, LDS pad of 16B after every 128B stored (-> 72-elem LDS rows).
// D# layout per CDNA5 ISA ch.8 (groups 0/1; groups 2/3 zero for 2-D tiles).
__device__ __forceinline__ void tdm_load_tile(unsigned lds_addr,
                                              unsigned long long gaddr,
                                              unsigned tensor_d0,
                                              unsigned stride0) {
  u32x4 g0;
  g0[0] = (unsigned)rfl(1u);                                  // count=1 (valid)
  g0[1] = (unsigned)rfl(lds_addr);                            // lds_addr bytes
  g0[2] = (unsigned)rfl((unsigned)(gaddr & 0xffffffffu));     // global[31:0]
  g0[3] = (unsigned)rfl((unsigned)((gaddr >> 32) & 0x1ffffffu) | (2u << 30)); // [56:32]|type=2
  i32x8 g1;
  // data_size=1(2B)<<16 | pad_enable<<20 | pad_interval=4(32 dw)<<22 | pad_amount=3(4 dw)<<25
  g1[0] = rfl((1u << 16) | (1u << 20) | (4u << 22) | (3u << 25));
  g1[1] = rfl((tensor_d0 & 0xffffu) << 16);                   // tdim0[15:0] @ [63:48]
  g1[2] = rfl(((tensor_d0 >> 16) & 0xffffu) | (64u << 16));   // tdim0[31:16] | tdim1=64
  g1[3] = rfl(64u << 16);                                     // tdim1 hi=0 | tile_dim0=64
  g1[4] = rfl(64u);                                           // tile_dim1=64, tile_dim2=0
  g1[5] = rfl(stride0);                                       // tensor_dim0_stride[31:0]
  g1[6] = rfl(0u);
  g1[7] = rfl(0u);
  i32x4 z4 = {0, 0, 0, 0};
  i32x8 z8 = {0, 0, 0, 0, 0, 0, 0, 0};
  // 6-arg form on this toolchain: (g0, g1, g2, g3, extra, cpol)
  __builtin_amdgcn_tensor_load_to_lds(g0, g1, z4, z4, z8, 0);
}
#endif

// ------------------------------ prep kernels -------------------------------
__global__ __launch_bounds__(256) void k_zero16(uint4* __restrict__ p, size_t n) {
  size_t i = (size_t)blockIdx.x * 256 + threadIdx.x;
  if (i < n) p[i] = make_uint4(0u, 0u, 0u, 0u);
}

__global__ __launch_bounds__(256) void k_cvt_x(const float* __restrict__ x,
                                               bf16* __restrict__ xb, int n) {
  int i = blockIdx.x * 256 + threadIdx.x;
  if (i < n) xb[i] = (bf16)x[i];
}

// w_qkv[d, c] (c = h*144 + which*48 + e) -> wqt[c, d]  (bf16, transposed)
__global__ __launch_bounds__(256) void k_cvt_wqkv(const float* __restrict__ w,
                                                  bf16* __restrict__ wt) {
  int i = blockIdx.x * 256 + threadIdx.x;
  if (i < C3 * D_) {
    int c = i / D_, d = i - c * D_;
    wt[i] = (bf16)w[(size_t)d * C3 + c];
  }
}

// w_o[e, h, d] -> wot[d, h*64 + e]  (bf16, e padded 48->64 with zeros)
__global__ __launch_bounds__(256) void k_cvt_wo(const float* __restrict__ w,
                                                bf16* __restrict__ wt) {
  int i = blockIdx.x * 256 + threadIdx.x;
  if (i < D_ * KOUT) {
    int d = i / KOUT, k = i - d * KOUT;
    int h = k >> 6, e = k & 63;
    float v = (e < PD_) ? w[((size_t)e * H_ + h) * D_ + d] : 0.0f;
    wt[i] = (bf16)v;
  }
}

// ------------------------------ QKV projection -----------------------------
// One wave per 16x16 output tile. M = 32 (b,p) * 1024 rows, Kdim = 384,
// Ncols = 1152.  Q scaled by 1/sqrt(PD); scatter into padded Q/K/V^T layouts.
__global__ __launch_bounds__(256) void k_qkv(const bf16* __restrict__ xb,
                                             const bf16* __restrict__ wt,
                                             const float* __restrict__ bqkv,
                                             bf16* __restrict__ Q,
                                             bf16* __restrict__ K,
                                             bf16* __restrict__ V) {
  const int lane = threadIdx.x & 31;
  const int lm   = lane & 15;
  size_t tile = (size_t)blockIdx.x * 8 + (threadIdx.x >> 5);
  int ct = (int)(tile % 72);  size_t t2 = tile / 72;
  int rt = (int)(t2 % 64);    int bp = (int)(t2 / 64);

  const bf16* arow = xb + ((size_t)bp * N_ + rt * 16 + lm) * D_;
  const bf16* brow = wt + (size_t)(ct * 16 + lm) * D_;
  f32x8 acc = {0.f, 0.f, 0.f, 0.f, 0.f, 0.f, 0.f, 0.f};
#pragma unroll
  for (int k = 0; k < D_; k += 32)
    acc = wmma_bf16(load_fragA(arow + k, lane), load_fragB(brow + k, lane), acc);

  const int c = ct * 16 + lm;              // 0..1151
  const int h = c / 144, r = c % 144;
  const int which = r / 48, e = r % 48;
  const float bias = bqkv[c];
  const float mul  = (which == 0) ? INV_SCALE : 1.0f;
  const size_t bph = (size_t)bp * H_ + h;
#pragma unroll
  for (int v = 0; v < 8; ++v) {
    int n = rt * 16 + v + 8 * (lane >> 4);
    bf16 val = (bf16)((acc[v] + bias) * mul);
    if (which == 0)      Q[(bph * N_ + n) * PDP + e] = val;
    else if (which == 1) K[(bph * N_ + n) * PDP + e] = val;
    else                 V[(bph * PDP + e) * N_ + n] = val;   // transposed
  }
}

// ------------------------------ flash attention ----------------------------
// Block = 4 waves = 64 query rows of one (b,p,h). K/V tiles double-buffered
// in LDS, streamed by the Tensor Data Mover (wave 0 issues, TENSORcnt waits).
__global__ __launch_bounds__(128) void k_attn(const bf16* __restrict__ Q,
                                              const bf16* __restrict__ K,
                                              const bf16* __restrict__ V,
                                              bf16* __restrict__ WV) {
  __shared__ __align__(16) bf16 Kl[2][64][72];     // K[j][e], 16B row pad
  __shared__ __align__(16) bf16 Vl[2][64][72];     // V^T[e][j]
  __shared__ __align__(16) bf16 Pl[4][16][72];     // per-wave P repack

  const int tid  = threadIdx.x;
  const int wave = tid >> 5;
  const int lane = tid & 31;
  const int lm   = lane & 15;
  const int bph  = blockIdx.x >> 4;                // 0..255
  const int nb   = (blockIdx.x & 15) * 64;         // query row block base

  const bf16* Qw = Q + ((size_t)bph * N_ + nb + wave * 16) * PDP;
  const bf16* Kg = K + (size_t)bph * N_ * PDP;
  const bf16* Vg = V + (size_t)bph * PDP * N_;

  const bf16* qrow = Qw + (size_t)lm * PDP;
  const bf16x16 aq0 = load_fragA(qrow,      lane); // e 0..31
  const bf16x16 aq1 = load_fragA(qrow + 32, lane); // e 32..63

  f32x8 acc[4];
  float mrun[8], lrun[8];
#pragma unroll
  for (int f = 0; f < 4; ++f) acc[f] = (f32x8){0.f,0.f,0.f,0.f,0.f,0.f,0.f,0.f};
#pragma unroll
  for (int v = 0; v < 8; ++v) { mrun[v] = -1e30f; lrun[v] = 0.f; }

#if HAS_TDM
  if (wave == 0) {    // prologue: stream tile 0 into buffer 0
    tdm_load_tile((unsigned)(uintptr_t)&Kl[0][0][0],
                  (unsigned long long)(uintptr_t)Kg, 64u, 64u);
    tdm_load_tile((unsigned)(uintptr_t)&Vl[0][0][0],
                  (unsigned long long)(uintptr_t)Vg, (unsigned)N_, (unsigned)N_);
  }
#endif

  for (int t = 0; t < N_ / 64; ++t) {
    const int jt  = t * 64;
    const int cur = t & 1;
    __syncthreads();   // all waves done reading the buffer we're about to fill
#if HAS_TDM
    if (wave == 0) {
      if (t + 1 < N_ / 64) {
        const int nxt = (t + 1) & 1;
        tdm_load_tile((unsigned)(uintptr_t)&Kl[nxt][0][0],
                      (unsigned long long)(uintptr_t)(Kg + (size_t)(jt + 64) * PDP),
                      64u, 64u);
        tdm_load_tile((unsigned)(uintptr_t)&Vl[nxt][0][0],
                      (unsigned long long)(uintptr_t)(Vg + (size_t)(jt + 64)),
                      (unsigned)N_, (unsigned)N_);
        __builtin_amdgcn_s_wait_tensorcnt((short)2);  // tile t's 2 loads done
      } else {
        __builtin_amdgcn_s_wait_tensorcnt((short)0);
      }
    }
#else
    if (jt + 64 < N_) {
      __builtin_prefetch((const void*)(Kg + (size_t)(jt + 64) * PDP), 0, 3);
      __builtin_prefetch((const void*)(Vg + jt + 64), 0, 3);
    }
#pragma unroll
    for (int it = 0; it < 4; ++it) {               // fallback: manual staging
      int idx = it * 128 + tid;                    // 0..511
      int row = idx >> 3, ch = (idx & 7) * 8;
      *(bf16x8*)&Kl[cur][row][ch] = *(const bf16x8*)(Kg + (size_t)(jt + row) * PDP + ch);
      *(bf16x8*)&Vl[cur][row][ch] = *(const bf16x8*)(Vg + (size_t)row * N_ + jt + ch);
    }
#endif
    __syncthreads();   // tile t resident in LDS for every wave

    // S = Q K^T  (scale pre-folded into Q)
    f32x8 s[4];
#pragma unroll
    for (int f = 0; f < 4; ++f) {
      const bf16* krow = &Kl[cur][f * 16 + lm][0];
      bf16x16 b0 = load_fragB(krow,      lane);
      bf16x16 b1 = load_fragB(krow + 32, lane);
      f32x8 c = {0.f,0.f,0.f,0.f,0.f,0.f,0.f,0.f};
      c = wmma_bf16(aq0, b0, c);
      c = wmma_bf16(aq1, b1, c);
      s[f] = c;
    }

    // online softmax: rows stripe across lane groups of 16 in the C layout
    float mnew[8], scal[8];
#pragma unroll
    for (int v = 0; v < 8; ++v) {
      float mx = fmaxf(fmaxf(s[0][v], s[1][v]), fmaxf(s[2][v], s[3][v]));
#pragma unroll
      for (int m = 1; m < 16; m <<= 1) mx = fmaxf(mx, __shfl_xor(mx, m, 16));
      mnew[v] = fmaxf(mrun[v], mx);
      scal[v] = __expf(mrun[v] - mnew[v]);
      mrun[v] = mnew[v];
    }
#pragma unroll
    for (int f = 0; f < 4; ++f) {
#pragma unroll
      for (int v = 0; v < 8; ++v) {
        float p = __expf(s[f][v] - mnew[v]);
        s[f][v] = p;
        Pl[wave][v + 8 * (lane >> 4)][f * 16 + lm] = (bf16)p;  // C->LDS repack
      }
    }
#pragma unroll
    for (int v = 0; v < 8; ++v) {
      float rs = s[0][v] + s[1][v] + s[2][v] + s[3][v];
#pragma unroll
      for (int m = 1; m < 16; m <<= 1) rs += __shfl_xor(rs, m, 16);
      lrun[v] = lrun[v] * scal[v] + rs;
    }
#pragma unroll
    for (int f = 0; f < 4; ++f)
#pragma unroll
      for (int v = 0; v < 8; ++v) acc[f][v] *= scal[v];

    // O += P V  (same-wave LDS ops are in order: stores precede these loads)
    const bf16* prow = &Pl[wave][lm][0];
    bf16x16 ap0 = load_fragA(prow,      lane);     // j 0..31
    bf16x16 ap1 = load_fragA(prow + 32, lane);     // j 32..63
#pragma unroll
    for (int f = 0; f < 4; ++f) {
      const bf16* vrow = &Vl[cur][f * 16 + lm][0]; // lane owns e = f*16+lm
      bf16x16 b0 = load_fragB(vrow,      lane);
      bf16x16 b1 = load_fragB(vrow + 32, lane);
      acc[f] = wmma_bf16(ap0, b0, acc[f]);
      acc[f] = wmma_bf16(ap1, b1, acc[f]);
    }
  }

  // normalize and write wv[b,p,n, h*64+e] (bf16) for the output projection
  const int    h  = bph & (H_ - 1);
  const size_t bp = (size_t)(bph >> 3);
#pragma unroll
  for (int v = 0; v < 8; ++v) {
    float inv = 1.0f / lrun[v];
    int n = nb + wave * 16 + v + 8 * (lane >> 4);
#pragma unroll
    for (int f = 0; f < 4; ++f)
      WV[(bp * N_ + n) * KOUT + h * 64 + f * 16 + lm] = (bf16)(acc[f][v] * inv);
  }
}

// ------------------------------ output projection --------------------------
// out[32768, 384] = WV[32768, 512] @ wot^T + b_o   (padded rows of wot are 0)
__global__ __launch_bounds__(256) void k_out(const bf16* __restrict__ WV,
                                             const bf16* __restrict__ wot,
                                             const float* __restrict__ bo,
                                             float* __restrict__ out) {
  const int lane = threadIdx.x & 31;
  const int lm   = lane & 15;
  size_t tile = (size_t)blockIdx.x * 8 + (threadIdx.x >> 5);
  int ct = (int)(tile % 24);
  int rt = (int)(tile / 24);                       // 0..2047
  const bf16* arow = WV  + ((size_t)rt * 16 + lm) * KOUT;
  const bf16* brow = wot + (size_t)(ct * 16 + lm) * KOUT;
  f32x8 acc = {0.f, 0.f, 0.f, 0.f, 0.f, 0.f, 0.f, 0.f};
#pragma unroll
  for (int k = 0; k < KOUT; k += 32)
    acc = wmma_bf16(load_fragA(arow + k, lane), load_fragB(brow + k, lane), acc);
  const int d = ct * 16 + lm;
  const float bias = bo[d];
#pragma unroll
  for (int v = 0; v < 8; ++v) {
    size_t row = (size_t)rt * 16 + v + 8 * (lane >> 4);
    out[row * D_ + d] = acc[v] + bias;
  }
}

// ------------------------------ launch -------------------------------------
extern "C" void kernel_launch(void* const* d_in, const int* in_sizes, int n_in,
                              void* d_out, int out_size, void* d_ws, size_t ws_size,
                              hipStream_t stream) {
  (void)in_sizes; (void)n_in; (void)out_size; (void)ws_size;
  const float* x     = (const float*)d_in[0];
  const float* w_qkv = (const float*)d_in[1];
  const float* b_qkv = (const float*)d_in[2];
  const float* w_o   = (const float*)d_in[3];
  const float* b_o   = (const float*)d_in[4];
  float* out = (float*)d_out;

  char* ws = (char*)d_ws;
  bf16* xb  = (bf16*)(ws + OFF_XB);
  bf16* wqt = (bf16*)(ws + OFF_WQT);
  bf16* wot = (bf16*)(ws + OFF_WOT);
  bf16* Q   = (bf16*)(ws + OFF_Q);
  bf16* K   = (bf16*)(ws + OFF_K);
  bf16* V   = (bf16*)(ws + OFF_VT);
  bf16* WV  = (bf16*)(ws + OFF_WV);

  // zero Q/K/V^T (covers the e=48..63 pads that k_qkv never writes)
  size_t nzero16 = (3 * SZ_QKV1) / 16;
  k_zero16<<<(unsigned)((nzero16 + 255) / 256), 256, 0, stream>>>((uint4*)(ws + OFF_Q), nzero16);

  const int nx = B_ * P_ * N_ * D_;
  k_cvt_x   <<<(nx + 255) / 256,        256, 0, stream>>>(x, xb, nx);
  k_cvt_wqkv<<<(C3 * D_ + 255) / 256,   256, 0, stream>>>(w_qkv, wqt);
  k_cvt_wo  <<<(D_ * KOUT + 255) / 256, 256, 0, stream>>>(w_o, wot);

  k_qkv <<<18432, 256, 0, stream>>>(xb, wqt, b_qkv, Q, K, V);
  k_attn<<< 4096, 128, 0, stream>>>(Q, K, V, WV);
  k_out <<< 6144, 256, 0, stream>>>(WV, wot, b_o, out);
}
